// CPUOptimizedMultiQueryAttention_62929860821689
// MI455X (gfx1250) — compile-verified, matrix operand order checked by reading, not run on hardware
//
#include <hip/hip_runtime.h>
#include <cstdint>

typedef __attribute__((ext_vector_type(16))) __bf16 v16bf;
typedef __attribute__((ext_vector_type(8)))  __bf16 v8bf;
typedef __attribute__((ext_vector_type(8)))  float  v8f;

#define BB  2
#define SS  2048
#define DD  1024
#define HH  16
#define DK  64

// ---- operand packing per CDNA5 16-bit A/B VGPR layout -----------------------
// lane l: row/col = l&15 ; kbase = 8*(l>>4); element e -> k = kbase + (e<8 ? e : 16+(e-8))
__device__ __forceinline__ v16bf pack_f32(const float* p) {
  v16bf r;
#pragma unroll
  for (int e = 0; e < 8; ++e) r[e] = (__bf16)p[e];
#pragma unroll
  for (int e = 0; e < 8; ++e) r[8 + e] = (__bf16)p[16 + e];
  return r;
}
__device__ __forceinline__ v16bf pack_bf16(const __bf16* p) {
  v8bf lo = *(const v8bf*)p;          // 16B aligned at all call sites
  v8bf hi = *(const v8bf*)(p + 16);
  return __builtin_shufflevector(lo, hi, 0,1,2,3,4,5,6,7,8,9,10,11,12,13,14,15);
}
__device__ __forceinline__ v8f wmma_bf16(v16bf a, v16bf b, v8f c) {
  return __builtin_amdgcn_wmma_f32_16x16x32_bf16(false, a, false, b, (short)0, c, false, false);
}
// 16-byte async global->LDS copy (ASYNCcnt-tracked, CDNA5)
__device__ __forceinline__ void async_copy16(const void* gptr, uint32_t lds_addr) {
  const uint64_t ga = (uint64_t)(size_t)gptr;
  asm volatile("global_load_async_to_lds_b128 %0, %1, off"
               :: "v"(lds_addr), "v"(ga) : "memory");
}

// ---- generic C[M,N] = A[M,K] @ W[N,K]^T + bias ------------------------------
// OUT_MODE 0: fp32 row-major.  1: bf16 row-major.
// OUT_MODE 2: bf16 head-split [B,H,S,dk].  3: bf16 transposed-per-batch [B,N,S].
template <bool A_BF16, int OUT_MODE>
__global__ __launch_bounds__(256) void gemm_wmma_nt(
    const void* __restrict__ A_, const float* __restrict__ W,
    const float* __restrict__ bias, void* __restrict__ out_,
    int M, int N, int K) {
  const int wave = threadIdx.x >> 5;
  const int lane = threadIdx.x & 31;
  const int hf   = lane >> 4;
  const int l16  = lane & 15;
  const int kb   = hf * 8;

  const int m0 = (blockIdx.x * 8 + wave) * 16;
  const int n0 = blockIdx.y * 16;

  v8f acc = {};
  const float* wrow = W + (size_t)(n0 + l16) * K + kb;
  if (A_BF16) {
    const __bf16* arow = (const __bf16*)A_ + (size_t)(m0 + l16) * K + kb;
    for (int k0 = 0; k0 < K; k0 += 32) {
      __builtin_prefetch(wrow + k0 + 256, 0, 1);
      acc = wmma_bf16(pack_bf16(arow + k0), pack_f32(wrow + k0), acc);
    }
  } else {
    const float* arow = (const float*)A_ + (size_t)(m0 + l16) * K + kb;
    for (int k0 = 0; k0 < K; k0 += 32) {
      __builtin_prefetch(wrow + k0 + 256, 0, 1);
      acc = wmma_bf16(pack_f32(arow + k0), pack_f32(wrow + k0), acc);
    }
  }

  const float bvv = bias[n0 + l16];
#pragma unroll
  for (int j = 0; j < 8; ++j) {
    const int m = m0 + j + 8 * hf;
    const int n = n0 + l16;
    const float ov = acc[j] + bvv;
    if (OUT_MODE == 0) {
      ((float*)out_)[(size_t)m * N + n] = ov;
    } else if (OUT_MODE == 1) {
      ((__bf16*)out_)[(size_t)m * N + n] = (__bf16)ov;
    } else if (OUT_MODE == 2) {
      const int b = m >> 11, s = m & (SS - 1);
      const int h = n >> 6,  d = n & (DK - 1);
      ((__bf16*)out_)[(((size_t)(b * HH + h)) * SS + s) * DK + d] = (__bf16)ov;
    } else {  // transposed per batch: [B, N, S]
      const int b = m >> 11, s = m & (SS - 1);
      ((__bf16*)out_)[((size_t)b * N + n) * SS + s] = (__bf16)ov;
    }
  }
}

// ---- flash MQA --------------------------------------------------------------
// Block = 8 waves; each wave owns a 16-query tile x dk=64 for one (b,h).
// K (32x64, row-major) and V^T (64x32, row-major from [B,dk,S]) tiles are
// double-buffered in LDS via async global->LDS copies.
__global__ __launch_bounds__(256) void mqa_flash(
    const __bf16* __restrict__ qh, const __bf16* __restrict__ kh,
    const __bf16* __restrict__ vhT, __bf16* __restrict__ ctx) {
  __shared__ __bf16 Kt[2][32 * DK];    // 2 x 4KB, K rows
  __shared__ __bf16 VtT[2][DK][32];    // 2 x 4KB, V^T rows (dk-major)
  __shared__ __bf16 Plds[8][16][32];   // per-wave P transpose staging (8KB)

  const int t    = threadIdx.x;
  const int wave = t >> 5;
  const int lane = t & 31;
  const int hf = lane >> 4, l16 = lane & 15, kb = hf * 8;
  const int qt = blockIdx.x * 8 + wave;
  const int h = blockIdx.y, b = blockIdx.z;

  const __bf16* qrow = qh + (((size_t)(b * HH + h) * SS) + qt * 16 + l16) * DK + kb;
  const v16bf qa0 = pack_bf16(qrow);        // k 0..31
  const v16bf qa1 = pack_bf16(qrow + 32);   // k 32..63

  const __bf16* Kb  = kh  + (size_t)b * SS * DK;
  const __bf16* VbT = vhT + (size_t)b * DK * SS;

  // 256 threads x 16B cover each 4KB tile.
  // K tile: contiguous rows.  V^T tile: 64 rows of 64B, row stride S*2 bytes.
  const int vr = t >> 2, vc = t & 3;
  auto issue_tile = [&](int kt, int buf) {
    const char* gk = (const char*)Kb + (size_t)kt * (32 * DK * 2) + t * 16;
    async_copy16(gk, (uint32_t)(size_t)(&Kt[buf][0]) + t * 16);
    const char* gv = (const char*)VbT + (((size_t)vr * SS) + (size_t)kt * 32) * 2 + vc * 16;
    async_copy16(gv, (uint32_t)(size_t)(&VtT[buf][vr][vc * 8]));
  };

  float mrun[8], lrun[8], alpha[8];
#pragma unroll
  for (int j = 0; j < 8; ++j) { mrun[j] = -__builtin_inff(); lrun[j] = 0.f; }
  v8f acc[4] = {};
  const float scale = 0.125f;  // 1/sqrt(64)
  const int NT = SS / 32;      // 64 key tiles

  issue_tile(0, 0);
  for (int kt = 0; kt < NT; ++kt) {
    const int cur = kt & 1;
    asm volatile("s_wait_asynccnt 0" ::: "memory");  // my tile-kt copies done
    __syncthreads();                                 // everyone's copies visible
    if (kt + 1 < NT) issue_tile(kt + 1, cur ^ 1);    // overlap next fill w/ compute

    // batch all four K B-operands so ds_load_b128s pipeline ahead of the WMMAs
    const __bf16* kr0 = &Kt[cur][(size_t)l16 * DK + kb];  // key cols 0..15
    const __bf16* kr1 = kr0 + 16 * DK;                    // key cols 16..31
    const v16bf kb00 = pack_bf16(kr0);
    const v16bf kb01 = pack_bf16(kr0 + 32);
    const v16bf kb10 = pack_bf16(kr1);
    const v16bf kb11 = pack_bf16(kr1 + 32);
    v8f s0 = {}, s1 = {};
    s0 = wmma_bf16(qa0, kb00, s0);
    s0 = wmma_bf16(qa1, kb01, s0);
    s1 = wmma_bf16(qa0, kb10, s1);
    s1 = wmma_bf16(qa1, kb11, s1);

    // online softmax: row r lives in VGPR r%8 on half (r/8); reduce across 16 lanes/half
#pragma unroll
    for (int j = 0; j < 8; ++j) {
      const float a0 = s0[j] * scale, a1 = s1[j] * scale;
      float cm = fmaxf(a0, a1);
#pragma unroll
      for (int msk = 1; msk < 16; msk <<= 1) cm = fmaxf(cm, __shfl_xor(cm, msk, 32));
      const float mnew = fmaxf(mrun[j], cm);
      alpha[j] = __expf(mrun[j] - mnew);
      const float p0 = __expf(a0 - mnew), p1 = __expf(a1 - mnew);
      float ts = p0 + p1;
#pragma unroll
      for (int msk = 1; msk < 16; msk <<= 1) ts += __shfl_xor(ts, msk, 32);
      lrun[j] = lrun[j] * alpha[j] + ts;
      mrun[j] = mnew;
      const int row = j + 8 * hf;
      Plds[wave][row][l16]      = (__bf16)p0;
      Plds[wave][row][16 + l16] = (__bf16)p1;
    }
#pragma unroll
    for (int g = 0; g < 4; ++g)
#pragma unroll
      for (int j = 0; j < 8; ++j) acc[g][j] *= alpha[j];

    asm volatile("s_wait_dscnt 0" ::: "memory");          // same-wave LDS RAW
    const v16bf pa = pack_bf16(&Plds[wave][l16][kb]);     // P as A operand (16x32)

    // V^T rows give contiguous B-operand chunks; batch loads, then 4 WMMAs
    v16bf vb0 = pack_bf16(&VtT[cur][0 * 16 + l16][kb]);
    v16bf vb1 = pack_bf16(&VtT[cur][1 * 16 + l16][kb]);
    v16bf vb2 = pack_bf16(&VtT[cur][2 * 16 + l16][kb]);
    v16bf vb3 = pack_bf16(&VtT[cur][3 * 16 + l16][kb]);
    acc[0] = wmma_bf16(pa, vb0, acc[0]);
    acc[1] = wmma_bf16(pa, vb1, acc[1]);
    acc[2] = wmma_bf16(pa, vb2, acc[2]);
    acc[3] = wmma_bf16(pa, vb3, acc[3]);
  }

  float inv[8];
#pragma unroll
  for (int j = 0; j < 8; ++j) inv[j] = 1.0f / lrun[j];
#pragma unroll
  for (int g = 0; g < 4; ++g)
#pragma unroll
    for (int j = 0; j < 8; ++j) {
      const int row = qt * 16 + j + 8 * hf;
      const int col = h * DK + g * 16 + l16;
      ctx[((size_t)b * SS + row) * DD + col] = (__bf16)(acc[g][j] * inv[j]);
    }
}

extern "C" void kernel_launch(void* const* d_in, const int* in_sizes, int n_in,
                              void* d_out, int out_size, void* d_ws, size_t ws_size,
                              hipStream_t stream) {
  const float* q  = (const float*)d_in[0];
  const float* k  = (const float*)d_in[1];
  const float* v  = (const float*)d_in[2];
  const float* Wq = (const float*)d_in[3];
  const float* bq = (const float*)d_in[4];
  const float* Wk = (const float*)d_in[5];
  const float* bk = (const float*)d_in[6];
  const float* Wv = (const float*)d_in[7];
  const float* bv = (const float*)d_in[8];
  const float* Wo = (const float*)d_in[9];
  const float* bo = (const float*)d_in[10];
  float* out = (float*)d_out;

  const size_t qh_bytes  = (size_t)BB * HH * SS * DK * 2;  // 8 MB
  const size_t kv_bytes  = (size_t)BB * SS * DK * 2;       // 0.5 MB
  const size_t ctx_bytes = (size_t)BB * SS * DD * 2;       // 8 MB
  if (ws_size < qh_bytes + 2 * kv_bytes + ctx_bytes) return;

  char* ws = (char*)d_ws;
  __bf16* qh   = (__bf16*)ws;
  __bf16* khp  = (__bf16*)(ws + qh_bytes);
  __bf16* vhpT = (__bf16*)(ws + qh_bytes + kv_bytes);      // [B, dk, S]
  __bf16* ctxb = (__bf16*)(ws + qh_bytes + 2 * kv_bytes);

  const dim3 blk(256);
  const int M = BB * SS;  // 4096

  // Q projection -> bf16 [B,H,S,dk]
  gemm_wmma_nt<false, 2><<<dim3(M / 16 / 8, DD / 16), blk, 0, stream>>>(
      q, Wq, bq, qh, M, DD, DD);
  // K projection -> bf16 [B,S,dk]
  gemm_wmma_nt<false, 1><<<dim3(M / 16 / 8, DK / 16), blk, 0, stream>>>(
      k, Wk, bk, khp, M, DK, DD);
  // V projection -> bf16 transposed [B,dk,S]
  gemm_wmma_nt<false, 3><<<dim3(M / 16 / 8, DK / 16), blk, 0, stream>>>(
      v, Wv, bv, vhpT, M, DK, DD);
  // flash attention -> bf16 ctx [B,S,D]
  mqa_flash<<<dim3(SS / 16 / 8, HH, BB), blk, 0, stream>>>(qh, khp, vhpT, ctxb);
  // output projection -> fp32 out
  gemm_wmma_nt<true, 0><<<dim3(M / 16 / 8, DD / 16), blk, 0, stream>>>(
      ctxb, Wo, bo, out, M, DD, DD);
}
